// QuantumAttention_65481071410515
// MI455X (gfx1250) — matrix operand order repaired and checked by reading, not compiled, not run on hardware
//
#include <hip/hip_runtime.h>

// ---------------------------------------------------------------------------
// MI455X (gfx1250) attention block, bf16 WMMA pipeline.
//   q,k,v = x@W* + b*  -> bf16      (wmma f32<-bf16, f32 bias add)
//   S = q k^T / 32     -> f32
//   P = softmax(S)     -> bf16
//   O = P v            -> bf16
//   out = O @ Wc + bc  -> f32
// GEMMs: A[M,K] x Bt[N,K] (B pre-transposed; K contiguous for both operands
// -> all LDS fragment reads are ds_load_b128). wave32, 8 waves/block,
// 128x128 block tile, 32x64 per-wave tile (1.5 ds_load_b128 per wmma).
// Global->LDS staging uses GLOBAL_LOAD_ASYNC_TO_LDS_B128 when the toolchain
// exposes it (ASYNCcnt path), else falls back to load+ds_store.
// ---------------------------------------------------------------------------

typedef __attribute__((ext_vector_type(16))) __bf16 v16bf;
typedef __attribute__((ext_vector_type(8)))  float v8f;
typedef __attribute__((ext_vector_type(4)))  int   v4i;

#define BMT 128          // block tile M
#define BNT 128          // block tile N
#define KTT 64           // block tile K
#define LDS_LD 72        // KTT + 8 bf16 pad -> 144B row stride (16B aligned)

#if defined(__has_builtin)
#  if __has_builtin(__builtin_amdgcn_global_load_async_to_lds_b128)
#    define USE_ASYNC_COPY 1
#  endif
#endif
#ifndef USE_ASYNC_COPY
#  define USE_ASYNC_COPY 0
#endif

__device__ inline void async_copy16(const void* g, void* l) {
#if USE_ASYNC_COPY
  // Param types per hipcc diagnostic: (int4 __device__*, ...). Pass typed
  // pointers; clang inserts the address-space casts for builtin args.
  __builtin_amdgcn_global_load_async_to_lds_b128((v4i*)g, (v4i*)l, 0, 0);
#else
  *(uint4*)l = *(const uint4*)g;
#endif
}

__device__ inline void wait_async_zero() {
#if USE_ASYNC_COPY
#if defined(__has_builtin) && __has_builtin(__builtin_amdgcn_s_wait_asynccnt)
  __builtin_amdgcn_s_wait_asynccnt(0);
#else
  asm volatile("s_wait_asynccnt 0x0" ::: "memory");
#endif
#endif
}

__device__ inline unsigned short f2bf(float f) {
  unsigned u = __float_as_uint(f);
  u += 0x7FFFu + ((u >> 16) & 1u);          // round-to-nearest-even
  return (unsigned short)(u >> 16);
}

union Frag16 { v16bf v; uint4 q[2]; };

// OUT_MODE: 0 = f32 row-major, 1 = bf16 row-major,
//           2 = bf16 transposed per-batch [b][n][t] (for Vt), S=2048,E=1024
template <int OUT_MODE, bool HAS_BIAS>
__global__ __launch_bounds__(256) void gemm_bf16_wmma(
    const unsigned short* __restrict__ A, int lda,
    const unsigned short* __restrict__ Bt, int ldb,
    void* __restrict__ Cout, int ldc,
    const float* __restrict__ bias, int K)
{
  __shared__ __align__(16) unsigned short Ash[BMT * LDS_LD];
  __shared__ __align__(16) unsigned short Bsh[BNT * LDS_LD];

  const int tid  = threadIdx.x;
  const int lane = tid & 31;
  const int wave = tid >> 5;
  const int wm = (wave & 3) * 32;           // 4 waves along M (32 rows each)
  const int wn = (wave >> 2) * 64;          // 2 waves along N (64 cols each)
  const int row0 = blockIdx.x * BMT;
  const int col0 = blockIdx.y * BNT;

  v8f acc[2][4] = {};

  for (int k0 = 0; k0 < K; k0 += KTT) {
    // Stage A tile (128x64 bf16): 1024 16B chunks, 4 per thread.
#pragma unroll
    for (int i = 0; i < 4; ++i) {
      int c = tid + i * 256;
      int r = c >> 3, kc = (c & 7) * 8;
      async_copy16(&A[(size_t)(row0 + r) * lda + k0 + kc],
                   &Ash[r * LDS_LD + kc]);
    }
    // Stage Bt tile (128x64 bf16): 1024 chunks, 4 per thread.
#pragma unroll
    for (int i = 0; i < 4; ++i) {
      int c = tid + i * 256;
      int r = c >> 3, kc = (c & 7) * 8;
      async_copy16(&Bt[(size_t)(col0 + r) * ldb + k0 + kc],
                   &Bsh[r * LDS_LD + kc]);
    }
    // Prefetch next K tile into cache while this one is consumed
    // (global_prefetch_b8, probe-verified on gfx1250).
    if (k0 + KTT < K) {
      int r = tid >> 1, kc = (tid & 1) * 32;
      __builtin_prefetch(&A[(size_t)(row0 + r) * lda + k0 + KTT + kc], 0, 0);
      __builtin_prefetch(&Bt[(size_t)(col0 + r) * ldb + k0 + KTT + kc], 0, 0);
    }
    wait_async_zero();
    __syncthreads();

#pragma unroll
    for (int kk = 0; kk < KTT; kk += 32) {
      // A fragment (16-bit 16x32): lanes 0-15 K 0-7 & 16-23, lanes 16-31
      // K 8-15 & 24-31  -> two ds_load_b128 per fragment.
      Frag16 fa[2], fb[4];
      const int am  = lane & 15;
      const int akb = (lane & 16) ? 8 : 0;
#pragma unroll
      for (int mi = 0; mi < 2; ++mi) {
        const unsigned short* ap = &Ash[(wm + mi * 16 + am) * LDS_LD + kk + akb];
        fa[mi].q[0] = *(const uint4*)ap;          // K = akb .. akb+7
        fa[mi].q[1] = *(const uint4*)(ap + 16);   // K = akb+16 .. akb+23
      }
      // B fragment (32x16): lanes 0-15 K 0-7 & 16-23, lanes 16-31 K 8-15 & 24-31.
      const int bn  = lane & 15;
      const int bkb = (lane & 16) ? 8 : 0;
#pragma unroll
      for (int ni = 0; ni < 4; ++ni) {
        const unsigned short* bp = &Bsh[(wn + ni * 16 + bn) * LDS_LD + kk + bkb];
        fb[ni].q[0] = *(const uint4*)bp;
        fb[ni].q[1] = *(const uint4*)(bp + 16);
      }
#pragma unroll
      for (int mi = 0; mi < 2; ++mi)
#pragma unroll
        for (int ni = 0; ni < 4; ++ni)
          acc[mi][ni] = __builtin_amdgcn_wmma_f32_16x16x32_bf16(
              false, fa[mi].v, false, fb[ni].v,
              (short)0, acc[mi][ni], false, false);
    }
    __syncthreads();
  }

  // C fragment: lane -> n = lane&15, rows m = ((lane>>4)<<3)+r, r=0..7
  const int cn = lane & 15;
  const int mh = (lane >> 4) << 3;
#pragma unroll
  for (int mi = 0; mi < 2; ++mi) {
#pragma unroll
    for (int ni = 0; ni < 4; ++ni) {
      const int gn  = col0 + wn + ni * 16 + cn;
      const int gm0 = row0 + wm + mi * 16 + mh;
      const float bv = HAS_BIAS ? bias[gn] : 0.0f;
      if (OUT_MODE == 0) {
        float* C = (float*)Cout;
#pragma unroll
        for (int r = 0; r < 8; ++r)
          C[(size_t)(gm0 + r) * ldc + gn] = acc[mi][ni][r] + bv;
      } else if (OUT_MODE == 1) {
        unsigned short* C = (unsigned short*)Cout;
#pragma unroll
        for (int r = 0; r < 8; ++r)
          C[(size_t)(gm0 + r) * ldc + gn] = f2bf(acc[mi][ni][r] + bv);
      } else {
        // Transposed per-batch store Vt[b][n][t]: 8 contiguous bf16 per lane
        // -> one global_store_b128.
        unsigned short* C = (unsigned short*)Cout;
        const int b = gm0 >> 11;          // S = 2048
        const int t = gm0 & 2047;
        __align__(16) unsigned short tmp[8];
#pragma unroll
        for (int r = 0; r < 8; ++r) tmp[r] = f2bf(acc[mi][ni][r] + bv);
        *(uint4*)&C[((size_t)b * 1024 + gn) * 2048 + t] = *(uint4*)tmp;
      }
    }
  }
}

// Row softmax over 2048 f32 -> bf16, with 1/sqrt(E)=1/32 pre-scale.
__global__ __launch_bounds__(256) void softmax_rows(
    const float* __restrict__ S, unsigned short* __restrict__ P)
{
  __shared__ float red[8];
  const int tid = threadIdx.x;
  const float* s = S + (size_t)blockIdx.x * 2048;
  float v[8], m = -3.4e38f;
#pragma unroll
  for (int i = 0; i < 8; ++i) {
    v[i] = s[tid + i * 256] * 0.03125f;
    m = fmaxf(m, v[i]);
  }
  for (int off = 16; off; off >>= 1) m = fmaxf(m, __shfl_xor(m, off, 32));
  if ((tid & 31) == 0) red[tid >> 5] = m;
  __syncthreads();
  float bm = red[0];
#pragma unroll
  for (int w = 1; w < 8; ++w) bm = fmaxf(bm, red[w]);
  __syncthreads();
  float sum = 0.f;
#pragma unroll
  for (int i = 0; i < 8; ++i) { v[i] = __expf(v[i] - bm); sum += v[i]; }
  for (int off = 16; off; off >>= 1) sum += __shfl_xor(sum, off, 32);
  if ((tid & 31) == 0) red[tid >> 5] = sum;
  __syncthreads();
  float bs = 0.f;
#pragma unroll
  for (int w = 0; w < 8; ++w) bs += red[w];
  const float inv = 1.0f / bs;
  unsigned short* p = P + (size_t)blockIdx.x * 2048;
#pragma unroll
  for (int i = 0; i < 8; ++i) p[tid + i * 256] = f2bf(v[i] * inv);
}

__global__ void convert_f32_bf16(const float* __restrict__ in,
                                 unsigned short* __restrict__ out, int n4)
{
  int i = blockIdx.x * blockDim.x + threadIdx.x;
  if (i >= n4) return;
  float4 f = ((const float4*)in)[i];
  __align__(8) unsigned short o[4] = { f2bf(f.x), f2bf(f.y), f2bf(f.z), f2bf(f.w) };
  ((uint2*)out)[i] = *(uint2*)o;
}

// W[k][n] f32 (1024x1024) -> WT[n][k] bf16, 32x32 LDS tiles.
__global__ __launch_bounds__(256) void transpose_convert_w(
    const float* __restrict__ W, unsigned short* __restrict__ WT)
{
  __shared__ unsigned short tile[32][33];
  const int n0 = blockIdx.x * 32, k0 = blockIdx.y * 32;
  const int tx = threadIdx.x, ty = threadIdx.y;   // 32 x 8
#pragma unroll
  for (int i = 0; i < 4; ++i) {
    int k = ty + i * 8;
    tile[k][tx] = f2bf(W[(size_t)(k0 + k) * 1024 + n0 + tx]);
  }
  __syncthreads();
#pragma unroll
  for (int i = 0; i < 4; ++i) {
    int n = ty + i * 8;
    WT[(size_t)(n0 + n) * 1024 + k0 + tx] = tile[tx][n];
  }
}

extern "C" void kernel_launch(void* const* d_in, const int* in_sizes, int n_in,
                              void* d_out, int out_size, void* d_ws, size_t ws_size,
                              hipStream_t stream)
{
  const float* x  = (const float*)d_in[0];
  const float* Wq = (const float*)d_in[1];
  const float* bq = (const float*)d_in[2];
  const float* Wk = (const float*)d_in[3];
  const float* bk = (const float*)d_in[4];
  const float* Wv = (const float*)d_in[5];
  const float* bv = (const float*)d_in[6];
  const float* Wc = (const float*)d_in[7];
  const float* bc = (const float*)d_in[8];
  float* out = (float*)d_out;

  const int B = 4, S = 2048, E = 1024;
  const int M = B * S;                       // 8192

  // Workspace layout (bytes), total ~112 MiB.
  char* ws = (char*)d_ws;
  size_t off = 0;
  unsigned short* xb  = (unsigned short*)(ws + off); off += (size_t)M * E * 2;   // 16Mi
  unsigned short* WqT = (unsigned short*)(ws + off); off += (size_t)E * E * 2;   // 2Mi
  unsigned short* WkT = (unsigned short*)(ws + off); off += (size_t)E * E * 2;
  unsigned short* WvT = (unsigned short*)(ws + off); off += (size_t)E * E * 2;
  unsigned short* WcT = (unsigned short*)(ws + off); off += (size_t)E * E * 2;
  unsigned short* Qb  = (unsigned short*)(ws + off); off += (size_t)M * E * 2;   // 16Mi
  unsigned short* Kb  = (unsigned short*)(ws + off); off += (size_t)M * E * 2;   // 16Mi
  unsigned short* Vt  = (unsigned short*)(ws + off); off += (size_t)M * E * 2;   // 16Mi [b][e][t]
  float*          Sc  = (float*)         (ws + off); off += (size_t)S * S * 4;   // 16Mi (per-batch reuse)
  unsigned short* Pb  = (unsigned short*)(ws + off); off += (size_t)S * S * 2;   //  8Mi (per-batch reuse)
  unsigned short* AO  = (unsigned short*)(ws + off); off += (size_t)M * E * 2;   // 16Mi

  // --- Phase 0: pack to bf16 -------------------------------------------------
  {
    int n4 = (M * E) / 4;
    convert_f32_bf16<<<(n4 + 255) / 256, 256, 0, stream>>>(x, xb, n4);
    dim3 g(32, 32), b(32, 8);
    transpose_convert_w<<<g, b, 0, stream>>>(Wq, WqT);
    transpose_convert_w<<<g, b, 0, stream>>>(Wk, WkT);
    transpose_convert_w<<<g, b, 0, stream>>>(Wv, WvT);
    transpose_convert_w<<<g, b, 0, stream>>>(Wc, WcT);
  }

  // --- Phase 1: Q/K/V projections (M=8192, N=1024, K=1024) -------------------
  {
    dim3 grid(M / BMT, E / BNT);   // 64 x 8
    gemm_bf16_wmma<1, true><<<grid, 256, 0, stream>>>(xb, E, WqT, E, Qb, E, bq, E);
    gemm_bf16_wmma<1, true><<<grid, 256, 0, stream>>>(xb, E, WkT, E, Kb, E, bk, E);
    gemm_bf16_wmma<2, true><<<grid, 256, 0, stream>>>(xb, E, WvT, E, Vt, E, bv, E);
  }

  // --- Phase 2: per-batch attention (scratch reused across batches) ----------
  for (int b = 0; b < B; ++b) {
    const unsigned short* Qp = Qb + (size_t)b * S * E;
    const unsigned short* Kp = Kb + (size_t)b * S * E;   // [t,e] == Bt[N,K] already
    const unsigned short* Vp = Vt + (size_t)b * E * S;   // [e,t] == Bt[N,K]
    unsigned short*       Op = AO + (size_t)b * S * E;

    dim3 gs(S / BMT, S / BNT);     // 16 x 16
    gemm_bf16_wmma<0, false><<<gs, 256, 0, stream>>>(Qp, E, Kp, E, Sc, S, nullptr, E);

    softmax_rows<<<S, 256, 0, stream>>>(Sc, Pb);

    dim3 gp(S / BMT, E / BNT);     // 16 x 8
    gemm_bf16_wmma<1, false><<<gp, 256, 0, stream>>>(Pb, S, Vp, S, Op, E, nullptr, S);
  }

  // --- Phase 3: output projection -> f32 d_out -------------------------------
  {
    dim3 grid(M / BMT, E / BNT);   // 64 x 8
    gemm_bf16_wmma<0, true><<<grid, 256, 0, stream>>>(AO, E, WcT, E, out, E, bc, E);
  }
}
